// TimeSeriesAttention_37160057045666
// MI455X (gfx1250) — matrix operand order repaired and checked by reading, not compiled, not run on hardware
//
#include <hip/hip_runtime.h>
#include <hip/hip_bf16.h>

// Problem constants (reference: B=4, T=4096, D=64)
#define BB 4
#define TT 4096
#define DD 64

typedef __attribute__((ext_vector_type(16))) __bf16 v16bf;
typedef __attribute__((ext_vector_type(8)))  float  v8f;

typedef __attribute__((ext_vector_type(4))) unsigned tdm_u4;
typedef __attribute__((ext_vector_type(8))) int      tdm_i8;
typedef __attribute__((ext_vector_type(4))) int      tdm_i4;

static __device__ __forceinline__ v8f wmma_bf16(v16bf a, v16bf b, v8f c) {
  return __builtin_amdgcn_wmma_f32_16x16x32_bf16(
      /*neg_a=*/false, a, /*neg_b=*/false, b,
      /*c_mod=*/(short)0, c, /*reuse_a=*/false, /*reuse_b=*/false);
}

// Async copy of one 16-byte chunk global->LDS (ASYNCcnt-tracked, no VGPRs).
// Low 32 bits of a generic LDS pointer are the LDS byte offset (ISA 10.2:
// LDS aperture addresses are truncated to addr[31:0]).
static __device__ __forceinline__ void async_cp16(void* lds_dst,
                                                  const void* gsrc) {
  unsigned lds = (unsigned)(size_t)lds_dst;
  unsigned long long ga = (unsigned long long)(size_t)gsrc;
  asm volatile("global_load_async_to_lds_b128 %0, %1, off"
               :: "v"(lds), "v"(ga) : "memory");
}
static __device__ __forceinline__ void async_wait0() {
  asm volatile("s_wait_asynccnt 0x0" ::: "memory");
}

// Tensor Data Mover: 1-D contiguous copy of n bf16 elements global->LDS.
// D# per ISA 8.3/8.4: group0 = {count=1, lds_addr, global_addr, type=2},
// group1 packs data_size=2B, tensor_dim0=tile_dim0=n, tensor_dim1=
// tile_dim1=1, stride0=n.  Issued once per wave (EXEC ignored), tracked by
// TENSORcnt.
static __device__ __forceinline__ void tdm_load_1d(void* lds_dst,
                                                   const void* gsrc, int n) {
  unsigned lds = (unsigned)(size_t)lds_dst;
  unsigned long long ga = (unsigned long long)(size_t)gsrc;
  tdm_u4 g0 = {
      0x1u,                                   // count=1 (valid user D#)
      lds,                                    // lds_addr
      (unsigned)ga,                           // global_addr[31:0]
      (unsigned)(ga >> 32) | (2u << 30)       // global_addr[56:32] | type=2
  };
  tdm_i8 g1 = {
      1 << 16,                                // data_size=1 (2 bytes)
      (n & 0xFFFF) << 16,                     // tensor_dim0[15:0]
      ((n >> 16) & 0xFFFF) | (1 << 16),       // tensor_dim0[31:16], td1=1
      (n & 0xFFFF) << 16,                     // tile_dim0
      1,                                      // tile_dim1=1, tile_dim2=0
      n,                                      // tensor_dim0_stride[31:0]
      0, 0                                    // stride0 hi, stride1
  };
  tdm_i4 g2 = {0, 0, 0, 0};
  tdm_i4 g3 = {0, 0, 0, 0};
#if defined(__clang_major__) && __clang_major__ >= 23
  tdm_i8 g4 = {0, 0, 0, 0, 0, 0, 0, 0};
  __builtin_amdgcn_tensor_load_to_lds(g0, g1, g2, g3, g4, 0);
#else
  __builtin_amdgcn_tensor_load_to_lds(g0, g1, g2, g3, 0);
#endif
}

// ---------------------------------------------------------------------------
// Kernel 1: per-row MLP  enc = tanh(softmax(x W1 + b1) W2 + b2)
// Also converts x to bf16.  One wave (32 lanes) per row; each lane owns
// output features {lane, lane+32}.  Softmax over 64 via wave32 shfl_xor.
// ---------------------------------------------------------------------------
__global__ __launch_bounds__(256) void encode_kernel(
    const float* __restrict__ x, const float* __restrict__ W1,
    const float* __restrict__ b1, const float* __restrict__ W2,
    const float* __restrict__ b2, __bf16* __restrict__ xb,
    __bf16* __restrict__ eb) {
  __shared__ float sW1[DD * DD];
  __shared__ float sW2[DD * DD];
  __shared__ float sx[8][DD];
  __shared__ float sh[8][DD];

  const int t = threadIdx.x;
  for (int i = t; i < DD * DD; i += 256) { sW1[i] = W1[i]; sW2[i] = W2[i]; }

  const int wave = t >> 5, lane = t & 31;
  const long row = (long)blockIdx.x * 8 + wave;  // row in [0, B*T)
  const float* xr = x + row * DD;
  sx[wave][lane]      = xr[lane];
  sx[wave][lane + 32] = xr[lane + 32];
  __syncthreads();

  // dense1
  float a0 = b1[lane], a1 = b1[lane + 32];
#pragma unroll 8
  for (int d = 0; d < DD; ++d) {
    float xv = sx[wave][d];
    a0 += xv * sW1[d * DD + lane];
    a1 += xv * sW1[d * DD + lane + 32];
  }
  // softmax over 64 features (2 per lane, wave32 tree reduce)
  float mx = fmaxf(a0, a1);
#pragma unroll
  for (int m = 16; m >= 1; m >>= 1) mx = fmaxf(mx, __shfl_xor(mx, m, 32));
  float e0 = __expf(a0 - mx), e1 = __expf(a1 - mx);
  float s = e0 + e1;
#pragma unroll
  for (int m = 16; m >= 1; m >>= 1) s += __shfl_xor(s, m, 32);
  const float inv = 1.0f / s;
  sh[wave][lane]      = e0 * inv;
  sh[wave][lane + 32] = e1 * inv;  // same-wave LDS ops are in order

  // dense2 + tanh
  float o0 = b2[lane], o1 = b2[lane + 32];
#pragma unroll 8
  for (int d = 0; d < DD; ++d) {
    float hv = sh[wave][d];
    o0 += hv * sW2[d * DD + lane];
    o1 += hv * sW2[d * DD + lane + 32];
  }
  eb[row * DD + lane]      = (__bf16)tanhf(o0);
  eb[row * DD + lane + 32] = (__bf16)tanhf(o1);
  xb[row * DD + lane]      = (__bf16)sx[wave][lane];
  xb[row * DD + lane + 32] = (__bf16)sx[wave][lane + 32];
}

// ---------------------------------------------------------------------------
// Kernel 2: per-row softmax stats of scores[b,j,i] = <x[b,j], enc[b,i]>.
// Block = 4 waves = 64 j rows.  Per iteration a 64x64 enc tile (8KB,
// contiguous) is DMA'd into LDS by the Tensor Data Mover (one
// tensor_load_to_lds issued by wave 0, double-buffered so tile t+1 streams
// under tile t's 8 WMMAs, completion via s_wait_tensorcnt).  Softmax stats
// are kept PER LANE (each lane only ever sees score column lane%16 of a
// tile) -> zero shuffles in the hot loop; the 16 lane-partials per row are
// merged with one log-sum-exp shuffle tree at the end.
// Writes m_j and 1/l_j.
// ---------------------------------------------------------------------------
__global__ __launch_bounds__(128) void stats_kernel(
    const __bf16* __restrict__ xb, const __bf16* __restrict__ eb,
    float* __restrict__ mrow, float* __restrict__ rlrow) {
  __shared__ __bf16 sEnc[2][64 * DD];  // 2 x 8KB double buffer

  const int t = threadIdx.x;
  const int wave = t >> 5, lane = t & 31;
  const int g = lane >> 4, m = lane & 15;
  const int bIdx = blockIdx.x >> 6;  // T/64 = 64 j-blocks per batch
  const int jblk = blockIdx.x & 63;
  const int j0 = jblk * 64 + wave * 16;

  // A fragments: rows of x (16x64 bf16 -> two 16x32 fragments).
  // A-layout: lane l (m=l%16, g=l/16): slot i -> K = g*16+i (matches B).
  const __bf16* xrow = xb + (long)(bIdx * TT + j0 + m) * DD;
  v16bf a0, a1;
#pragma unroll
  for (int i = 0; i < 8; ++i) {
    a0[i]     = xrow[g * 16 + i];
    a0[8 + i] = xrow[g * 16 + 8 + i];
    a1[i]     = xrow[32 + g * 16 + i];
    a1[8 + i] = xrow[40 + g * 16 + i];
  }

  // Per-lane online softmax partials (columns {64*it + isub*16 + m}).
  float mr[8], lr[8];
#pragma unroll
  for (int r = 0; r < 8; ++r) { mr[r] = -3.0e38f; lr[r] = 0.0f; }

  const __bf16* ebase = eb + (long)bIdx * TT * DD;

  if (wave == 0) {
    tdm_load_1d(&sEnc[0][0], ebase, 64 * DD);
    __builtin_amdgcn_s_wait_tensorcnt(0);
  }
  __syncthreads();

  for (int it = 0; it < TT / 64; ++it) {
    const int cur = it & 1;
    if (wave == 0 && it + 1 < TT / 64)  // TDM prefetch of next tile
      tdm_load_1d(&sEnc[cur ^ 1][0], ebase + (long)(it + 1) * 64 * DD,
                  64 * DD);
    const __bf16* sb = &sEnc[cur][0];
#pragma unroll
    for (int isub = 0; isub < 4; ++isub) {
      // B fragment from LDS: lane holds N=m, K=g*16+idx -> contiguous 32B.
      const __bf16* erow = sb + (isub * 16 + m) * DD + g * 16;
      v16bf b0 = *(const v16bf*)erow;
      v16bf b1 = *(const v16bf*)(erow + 32);
      v8f c = {};
      c = wmma_bf16(a0, b0, c);
      c = wmma_bf16(a1, b1, c);
      // Per-lane online update, no cross-lane traffic.
#pragma unroll
      for (int r = 0; r < 8; ++r) {
        float sv = c[r];
        float mn = fmaxf(mr[r], sv);
        lr[r] = lr[r] * __expf(mr[r] - mn) + __expf(sv - mn);
        mr[r] = mn;
      }
    }
    if (wave == 0) __builtin_amdgcn_s_wait_tensorcnt(0);
    __syncthreads();
  }

  // Merge the 16 lane-partials per row (stays within each 16-lane half).
#pragma unroll
  for (int r = 0; r < 8; ++r) {
    float mm = mr[r], ll = lr[r];
#pragma unroll
    for (int mask = 1; mask <= 8; mask <<= 1) {
      float mo = __shfl_xor(mm, mask, 32);
      float lo = __shfl_xor(ll, mask, 32);
      float mn = fmaxf(mm, mo);
      ll = ll * __expf(mm - mn) + lo * __expf(mo - mn);
      mm = mn;
    }
    mr[r] = mm;
    lr[r] = ll;
  }
  if (m == 0) {
#pragma unroll
    for (int r = 0; r < 8; ++r) {
      long j = (long)bIdx * TT + j0 + g * 8 + r;
      mrow[j] = mr[r];
      rlrow[j] = 1.0f / lr[r];
    }
  }
}

// ---------------------------------------------------------------------------
// Kernel 3: out[b,i,:] = sum_j exp(S[j,i]-m_j)/l_j * x[b,j,:]
// Each wave owns a 16-row i-tile (all 64 d columns in 4 accumulators).
// Per 32-wide j-tile: recompute 2 score tiles via WMMA, convert to P in
// registers (the score tile's C layout IS the A layout of P^T -> no
// shuffles), then 4 more WMMAs chain-accumulate out.  Softmax stats are
// async-copied to LDS once; x j-tile staged transposed for contiguous
// ds-load B fragments.
// ---------------------------------------------------------------------------
__global__ __launch_bounds__(128) void output_kernel(
    const __bf16* __restrict__ xb, const __bf16* __restrict__ eb,
    const float* __restrict__ mrow, const float* __restrict__ rlrow,
    float* __restrict__ out) {
  __shared__ float sm[TT];         // 16KB
  __shared__ float srl[TT];        // 16KB
  __shared__ __bf16 xT[DD * 32];   // xT[d][jj], 4KB

  const int t = threadIdx.x;
  const int wave = t >> 5, lane = t & 31, g = lane >> 4;
  const int bIdx = blockIdx.x >> 6, iblk = blockIdx.x & 63;
  const int i0 = iblk * 64 + wave * 16;

  // Async preload of this batch's softmax stats (2 x 16KB contiguous).
  {
    const char* s0 = (const char*)(mrow + (long)bIdx * TT);
    const char* s1 = (const char*)(rlrow + (long)bIdx * TT);
    char* d0 = (char*)&sm[0];
    char* d1 = (char*)&srl[0];
#pragma unroll
    for (int q = 0; q < 8; ++q) {
      int c = (t + 128 * q) * 16;  // byte offset within 16KB
      async_cp16(d0 + c, s0 + c);
      async_cp16(d1 + c, s1 + c);
    }
    async_wait0();
  }

  // enc B fragments for this wave's i-tile (fixed across the j loop).
  const __bf16* er = eb + (long)(bIdx * TT + i0 + (lane & 15)) * DD + g * 16;
  v16bf be0 = *(const v16bf*)er;
  v16bf be1 = *(const v16bf*)(er + 32);

  v8f acc[4];
#pragma unroll
  for (int d = 0; d < 4; ++d) acc[d] = (v8f){};
  __syncthreads();

  const __bf16* xbase = xb + (long)bIdx * TT * DD;
  for (int j0 = 0; j0 < TT; j0 += 32) {
    __syncthreads();  // previous tile's fragment reads complete
    // Stage x[j0..j0+31][0..63] transposed: xT[d][jj].  Dword global
    // loads (2 bf16) + two b16 LDS stores per element pair.
    for (int k = t; k < 32 * 32; k += 128) {
      int dp = k & 31;   // d pair -> d = 2*dp, 2*dp+1
      int jj = k >> 5;   // 0..31
      unsigned u = *(const unsigned*)(xbase + (long)(j0 + jj) * DD + 2 * dp);
      __bf16 b2[2];
      __builtin_memcpy(b2, &u, 4);
      xT[(2 * dp) * 32 + jj]     = b2[0];
      xT[(2 * dp + 1) * 32 + jj] = b2[1];
    }
    __syncthreads();

    // Score tiles S0 (j rows j0..j0+15), S1 (j0+16..j0+31)
    const __bf16* xr0 = xbase + (long)(j0 + (lane & 15)) * DD;
    const __bf16* xr1 = xr0 + 16 * DD;
    v16bf ax0, ax1, ay0, ay1;
#pragma unroll
    for (int i = 0; i < 8; ++i) {
      ax0[i] = xr0[g * 16 + i];      ax0[8 + i] = xr0[g * 16 + 8 + i];
      ax1[i] = xr0[32 + g * 16 + i]; ax1[8 + i] = xr0[40 + g * 16 + i];
      ay0[i] = xr1[g * 16 + i];      ay0[8 + i] = xr1[g * 16 + 8 + i];
      ay1[i] = xr1[32 + g * 16 + i]; ay1[8 + i] = xr1[40 + g * 16 + i];
    }
    v8f s0 = {}, s1 = {};
    s0 = wmma_bf16(ax0, be0, s0);
    s0 = wmma_bf16(ax1, be1, s0);
    s1 = wmma_bf16(ay0, be0, s1);
    s1 = wmma_bf16(ay1, be1, s1);

    // P^T A-fragment: C slot r of S0 holds S[j=g*8+r][i=lane%16] which is
    // exactly A slot r (K=g*8+r) for M=i=lane%16; S1 fills slots 8..15.
    v16bf apt;
#pragma unroll
    for (int r = 0; r < 8; ++r) {
      int ja = j0 + g * 8 + r;
      apt[r] = (__bf16)(__expf(s0[r] - sm[ja]) * srl[ja]);
      int jb = j0 + 16 + g * 8 + r;
      apt[8 + r] = (__bf16)(__expf(s1[r] - sm[jb]) * srl[jb]);
    }

    // out_tile += P^T (16x32) @ x_tile (32x16 per d-tile), 4 d-tiles.
#pragma unroll
    for (int dt = 0; dt < 4; ++dt) {
      const __bf16* lp = &xT[(dt * 16 + (lane & 15)) * 32 + g * 8];
      v16bf bx;
#pragma unroll
      for (int i = 0; i < 8; ++i) { bx[i] = lp[i]; bx[8 + i] = lp[16 + i]; }
      acc[dt] = wmma_bf16(apt, bx, acc[dt]);
    }
  }

  // Store: C layout -> out[b, i0 + g*8 + r, dt*16 + lane%16]
#pragma unroll
  for (int dt = 0; dt < 4; ++dt)
#pragma unroll
    for (int r = 0; r < 8; ++r)
      out[(long)(bIdx * TT + i0 + g * 8 + r) * DD + dt * 16 + (lane & 15)] =
          acc[dt][r];
}

extern "C" void kernel_launch(void* const* d_in, const int* in_sizes, int n_in,
                              void* d_out, int out_size, void* d_ws,
                              size_t ws_size, hipStream_t stream) {
  const float* x  = (const float*)d_in[0];
  const float* W1 = (const float*)d_in[1];
  const float* b1 = (const float*)d_in[2];
  const float* W2 = (const float*)d_in[3];
  const float* b2 = (const float*)d_in[4];
  float* out = (float*)d_out;

  // Workspace layout: xb (2MB bf16) | eb (2MB bf16) | m (64KB f32) | rl (64KB)
  char* ws = (char*)d_ws;
  __bf16* xb = (__bf16*)ws;
  __bf16* eb = (__bf16*)(ws + (size_t)BB * TT * DD * 2);
  float* mrow = (float*)(ws + (size_t)BB * TT * DD * 4);
  float* rlrow = mrow + (size_t)BB * TT;

  // 1) MLP encode + bf16 conversion: one wave per row, 8 rows/block.
  encode_kernel<<<BB * TT / 8, 256, 0, stream>>>(x, W1, b1, W2, b2, xb, eb);
  // 2) Row softmax stats (TDM-staged enc tiles, per-lane online softmax).
  stats_kernel<<<BB * (TT / 64), 128, 0, stream>>>(xb, eb, mrow, rlrow);
  // 3) Recompute scores + transposed apply, WMMA end to end.
  output_kernel<<<BB * (TT / 64), 128, 0, stream>>>(xb, eb, mrow, rlrow, out);
}
